// DrugEncoder_69071664054692
// MI455X (gfx1250) — matrix operand order repaired and checked by reading, not compiled, not run on hardware
//
#include <hip/hip_runtime.h>
#include <hip/hip_bf16.h>
#include <stdint.h>

// ---------------------------------------------------------------------------
// GNN encoder: GraphNorm -> GCNConv(64->128) -> ReLU -> GraphNorm ->
//              GCNConv(128->64) -> global_mean_pool.
// Roofline: ~1.5-2 GB of edge gather/scatter traffic vs 1.6 GFLOP dense math
// -> memory bound; GEMMs ride the wave32 WMMA pipe (bf16 in, f32 acc) with
// fragment-packed weights, vectorized A loads, constant-stride epilogue.
// ---------------------------------------------------------------------------

typedef __attribute__((ext_vector_type(16))) __bf16 v16bf;
typedef __attribute__((ext_vector_type(8)))  float  v8f;

#define GN_EPS 1e-5f
#define NGRAPH 256

static __device__ __forceinline__ __bf16 f2bf(float f) {
  // round-to-nearest-even fp32 -> bf16 (bit manipulation; no fp-cvt dependency)
  uint32_t u = __builtin_bit_cast(uint32_t, f);
  uint32_t r = (u + 0x7FFFu + ((u >> 16) & 1u)) >> 16;
  uint16_t s = (uint16_t)r;
  return __builtin_bit_cast(__bf16, s);
}

// ---------------- elementwise / segment kernels (float4 chunks) ------------

__global__ void k_zero(float* __restrict__ p, long long n) {
  long long i = (long long)blockIdx.x * blockDim.x + threadIdx.x;
  if (i < n) p[i] = 0.0f;
}

__global__ void k_count(const int* __restrict__ batch, float* __restrict__ cnt, int N) {
  int i = blockIdx.x * blockDim.x + threadIdx.x;
  if (i < N) atomicAdd(&cnt[batch[i]], 1.0f);
}

__global__ void k_segsum4(const float* __restrict__ x, const int* __restrict__ batch,
                          float* __restrict__ sum, int N, int D) {
  int C4 = D >> 2;
  long long i = (long long)blockIdx.x * blockDim.x + threadIdx.x;
  if (i < (long long)N * C4) {
    int n = (int)(i / C4), d4 = ((int)(i % C4)) << 2;
    int g = batch[n];
    float4 v = *(const float4*)(x + (size_t)n * D + d4);
    float* s = sum + (size_t)g * D + d4;
    atomicAdd(s + 0, v.x);
    atomicAdd(s + 1, v.y);
    atomicAdd(s + 2, v.z);
    atomicAdd(s + 3, v.w);
  }
}

__global__ void k_center_var4(const float* __restrict__ x, const int* __restrict__ batch,
                              const float* __restrict__ sum, const float* __restrict__ cnt,
                              const float* __restrict__ ms, float* __restrict__ out,
                              float* __restrict__ var, int N, int D) {
  int C4 = D >> 2;
  long long i = (long long)blockIdx.x * blockDim.x + threadIdx.x;
  if (i < (long long)N * C4) {
    int n = (int)(i / C4), d4 = ((int)(i % C4)) << 2;
    int g = batch[n];
    float c = fmaxf(cnt[g], 1.0f);
    float4 s = *(const float4*)(sum + (size_t)g * D + d4);
    float4 m = *(const float4*)(ms + d4);
    float4 xv = *(const float4*)(x + (size_t)n * D + d4);
    float4 v;
    v.x = xv.x - m.x * (s.x / c);
    v.y = xv.y - m.y * (s.y / c);
    v.z = xv.z - m.z * (s.z / c);
    v.w = xv.w - m.w * (s.w / c);
    *(float4*)(out + (size_t)n * D + d4) = v;
    float* vp = var + (size_t)g * D + d4;
    atomicAdd(vp + 0, v.x * v.x);
    atomicAdd(vp + 1, v.y * v.y);
    atomicAdd(vp + 2, v.z * v.z);
    atomicAdd(vp + 3, v.w * v.w);
  }
}

__global__ void k_gn_final4(float* __restrict__ out, const int* __restrict__ batch,
                            const float* __restrict__ var, const float* __restrict__ cnt,
                            const float* __restrict__ w, const float* __restrict__ b,
                            int N, int D) {
  int C4 = D >> 2;
  long long i = (long long)blockIdx.x * blockDim.x + threadIdx.x;
  if (i < (long long)N * C4) {
    int n = (int)(i / C4), d4 = ((int)(i % C4)) << 2;
    int g = batch[n];
    float c = fmaxf(cnt[g], 1.0f);
    float4 vv = *(const float4*)(var + (size_t)g * D + d4);
    float4 wv = *(const float4*)(w + d4);
    float4 bv = *(const float4*)(b + d4);
    float4 ov = *(const float4*)(out + (size_t)n * D + d4);
    float4 r;
    r.x = wv.x * (ov.x * rsqrtf(vv.x / c + GN_EPS)) + bv.x;
    r.y = wv.y * (ov.y * rsqrtf(vv.y / c + GN_EPS)) + bv.y;
    r.z = wv.z * (ov.z * rsqrtf(vv.z / c + GN_EPS)) + bv.z;
    r.w = wv.w * (ov.w * rsqrtf(vv.w / c + GN_EPS)) + bv.w;
    *(float4*)(out + (size_t)n * D + d4) = r;
  }
}

__global__ void k_deg_init(float* __restrict__ deg, int N) {
  int i = blockIdx.x * blockDim.x + threadIdx.x;
  if (i < N) deg[i] = 1.0f;   // self-loop
}

__global__ void k_deg_count(const int* __restrict__ ei, float* __restrict__ deg, int E) {
  int e = blockIdx.x * blockDim.x + threadIdx.x;
  if (e < E) atomicAdd(&deg[ei[E + e]], 1.0f);   // targets = ei[1][:]
}

__global__ void k_dis(const float* __restrict__ deg, float* __restrict__ dis, int N) {
  int i = blockIdx.x * blockDim.x + threadIdx.x;
  if (i < N) dis[i] = rsqrtf(deg[i]);            // deg >= 1 always (self-loops)
}

// self-loop term + bias initializes the accumulator (non-atomic full write)
__global__ void k_agg_init4(const float* __restrict__ h, const float* __restrict__ dis,
                            const float* __restrict__ b, float* __restrict__ out,
                            int N, int D) {
  int C4 = D >> 2;
  long long i = (long long)blockIdx.x * blockDim.x + threadIdx.x;
  if (i < (long long)N * C4) {
    int n = (int)(i / C4), d4 = ((int)(i % C4)) << 2;
    float s = dis[n];
    float s2 = s * s;
    float4 hv = *(const float4*)(h + (size_t)n * D + d4);
    float4 bv = *(const float4*)(b + d4);
    float4 r;
    r.x = hv.x * s2 + bv.x;
    r.y = hv.y * s2 + bv.y;
    r.z = hv.z * s2 + bv.z;
    r.w = hv.w * s2 + bv.w;
    *(float4*)(out + (size_t)n * D + d4) = r;
  }
}

__global__ void k_agg_edges4(const float* __restrict__ h, const int* __restrict__ ei,
                             const float* __restrict__ dis, float* __restrict__ out,
                             int E, int D) {
  int C4 = D >> 2;
  long long i = (long long)blockIdx.x * blockDim.x + threadIdx.x;
  if (i < (long long)E * C4) {
    int e = (int)(i / C4), d4 = ((int)(i % C4)) << 2;
    int r = ei[e], c = ei[E + e];
    float w = dis[r] * dis[c];
    float4 hv = *(const float4*)(h + (size_t)r * D + d4);
    float* o = out + (size_t)c * D + d4;
    atomicAdd(o + 0, hv.x * w);
    atomicAdd(o + 1, hv.y * w);
    atomicAdd(o + 2, hv.z * w);
    atomicAdd(o + 3, hv.w * w);
  }
}

__global__ void k_relu4(float* __restrict__ p, long long n4) {
  long long i = (long long)blockIdx.x * blockDim.x + threadIdx.x;
  if (i < n4) {
    float4 v = *(const float4*)(p + i * 4);
    v.x = fmaxf(v.x, 0.0f);
    v.y = fmaxf(v.y, 0.0f);
    v.z = fmaxf(v.z, 0.0f);
    v.w = fmaxf(v.w, 0.0f);
    *(float4*)(p + i * 4) = v;
  }
}

__global__ void k_pool_div(const float* __restrict__ sum, const float* __restrict__ cnt,
                           float* __restrict__ out, int D) {
  int i = blockIdx.x * blockDim.x + threadIdx.x;
  if (i < NGRAPH * D) {
    int g = i / D;
    out[i] = sum[i] / fmaxf(cnt[g], 1.0f);
  }
}

// ---------------- weight pre-pack into WMMA B-fragment layout --------------
// Bp[((tn*KB + kb)*32 + lane)*16 + j] = bf16(W[(kb*32 + (lane>>4)*16 + j)*Nc
//                                              + tn*16 + (lane&15)])
// so each lane's 16-element fragment is a contiguous, 32B-aligned chunk.

__global__ void k_pack_b(const float* __restrict__ W, __bf16* __restrict__ Bp,
                         int K, int Nc) {
  int i = blockIdx.x * blockDim.x + threadIdx.x;
  if (i < K * Nc) {
    int KB = K >> 5;
    int j = i & 15;
    int lane = (i >> 4) & 31;
    int rest = i >> 9;                // tn*KB + kb
    int kb = rest % KB;
    int tn = rest / KB;
    int half = lane >> 4, l15 = lane & 15;
    int kk = (kb << 5) + half * 16 + j;
    int n = (tn << 4) + l15;
    Bp[i] = f2bf(W[kk * Nc + n]);
  }
}

// ---------------- WMMA GEMM: C[M,Nc] = A[M,K] @ B[K,Nc] --------------------
// Templated on K, Nc (fully unrolled, constant strides) and TNT (N-tiles per
// wave; the A fragment is converted once per k-step and reused across TNT
// WMMAs). A/B k contiguous (16*half + s); C layout per ISA 7.12.2. Loads unconditional (row clamped,
// tail masked by multiply); stores branch-free when the M-tile is full.

template <int K, int Nc, int TNT>
__global__ void k_gemm_wmma(const float* __restrict__ A, const __bf16* __restrict__ Bp,
                            float* __restrict__ C, int M) {
  constexpr int KB = K / 32;
  int lane = threadIdx.x & 31;
  int wave = (int)(((long long)blockIdx.x * blockDim.x + threadIdx.x) >> 5);
  constexpr int groups_n = (Nc >> 4) / TNT;
  int tiles_m = (M + 15) >> 4;
  if (wave >= tiles_m * groups_n) return;          // uniform per wave
  int tm = wave / groups_n;
  int tg = wave % groups_n;
  int half = lane >> 4, l15 = lane & 15;

  int row = tm * 16 + l15;                         // A: M = l15 in both halves
  float fm = row < M ? 1.0f : 0.0f;                // zero-mask for tail rows
  int rc = row < M ? row : (M - 1);                // clamped (always-valid) row
  const float* arow = A + (size_t)rc * K;

  v8f acc[TNT];
#pragma unroll
  for (int t = 0; t < TNT; ++t) acc[t] = (v8f){};

#pragma unroll
  for (int kb = 0; kb < KB; ++kb) {
    // A 16x32 bf16 fragment: half 0 -> K = kb*32+{0..15}; half 1 -> kb*32+{16..31}
    int ka = kb * 32 + half * 16;
    float4 a0 = *(const float4*)(arow + ka);
    float4 a1 = *(const float4*)(arow + ka + 4);
    float4 a2 = *(const float4*)(arow + ka + 8);
    float4 a3 = *(const float4*)(arow + ka + 12);
    v16bf a;
    a[0]  = f2bf(a0.x * fm); a[1]  = f2bf(a0.y * fm);
    a[2]  = f2bf(a0.z * fm); a[3]  = f2bf(a0.w * fm);
    a[4]  = f2bf(a1.x * fm); a[5]  = f2bf(a1.y * fm);
    a[6]  = f2bf(a1.z * fm); a[7]  = f2bf(a1.w * fm);
    a[8]  = f2bf(a2.x * fm); a[9]  = f2bf(a2.y * fm);
    a[10] = f2bf(a2.z * fm); a[11] = f2bf(a2.w * fm);
    a[12] = f2bf(a3.x * fm); a[13] = f2bf(a3.y * fm);
    a[14] = f2bf(a3.z * fm); a[15] = f2bf(a3.w * fm);
#pragma unroll
    for (int t = 0; t < TNT; ++t) {
      int tn = tg * TNT + t;
      v16bf bb = *(const v16bf*)(Bp + (((size_t)tn * KB + kb) * 32 + lane) * 16);
      acc[t] = __builtin_amdgcn_wmma_f32_16x16x32_bf16(
          /*neg_a=*/false, a, /*neg_b=*/false, bb,
          /*c_mod=*/(short)0, acc[t], /*reuse_a=*/false, /*reuse_b=*/false);
    }
  }

  // C 16x16 f32: VGPR r -> M = half*8 + r, N = l15. Constant row stride ->
  // immediate-offset stores from one per-lane base pointer.
  int cm0 = tm * 16 + half * 8;
  float* cbase = C + (size_t)cm0 * Nc + tg * (TNT * 16) + l15;
  if (cm0 + 8 <= M) {                              // full tile: branch-free
#pragma unroll
    for (int t = 0; t < TNT; ++t)
#pragma unroll
      for (int r = 0; r < 8; ++r)
        cbase[(size_t)r * Nc + t * 16] = acc[t][r];
  } else {                                         // M tail (never hit if M%16==0)
#pragma unroll
    for (int t = 0; t < TNT; ++t)
#pragma unroll
      for (int r = 0; r < 8; ++r)
        if (cm0 + r < M) cbase[(size_t)r * Nc + t * 16] = acc[t][r];
  }
}

// ---------------------------------------------------------------------------

extern "C" void kernel_launch(void* const* d_in, const int* in_sizes, int n_in,
                              void* d_out, int out_size, void* d_ws, size_t ws_size,
                              hipStream_t stream) {
  const float* x     = (const float*)d_in[0];
  const int*   ei    = (const int*)d_in[1];
  const int*   batch = (const int*)d_in[2];
  const float* gn1w  = (const float*)d_in[3];
  const float* gn1b  = (const float*)d_in[4];
  const float* gn1ms = (const float*)d_in[5];
  const float* W1    = (const float*)d_in[6];
  const float* b1    = (const float*)d_in[7];
  const float* gn2w  = (const float*)d_in[8];
  const float* gn2b  = (const float*)d_in[9];
  const float* gn2ms = (const float*)d_in[10];
  const float* W2    = (const float*)d_in[11];
  const float* b2    = (const float*)d_in[12];
  float* out = (float*)d_out;

  const int DI = 64, DH = 128, DO = 64;
  const int N = in_sizes[0] / DI;
  const int E = in_sizes[1] / 2;

  // ---- workspace carve-up (floats, 32B-aligned chunks) ----
  float* ws = (float*)d_ws;
  size_t off = 0;
  auto align8 = [&off]() { off = (off + 7) & ~(size_t)7; };
  float* deg = ws + off; off += N; align8();
  float* dis = ws + off; off += N; align8();
  float* cnt = ws + off; off += NGRAPH; align8();
  float* sum = ws + off; off += (size_t)NGRAPH * DH; align8();
  float* var = ws + off; off += (size_t)NGRAPH * DH; align8();
  float* gn1 = ws + off; off += (size_t)N * DI; align8();
  float* h1  = ws + off; off += (size_t)N * DH; align8();
  float* ag1 = ws + off; off += (size_t)N * DH; align8();
  float* gn2 = ws + off; off += (size_t)N * DH; align8();
  float* h2  = ws + off; off += (size_t)N * DO; align8();
  float* ag2 = ws + off; off += (size_t)N * DO; align8();
  __bf16* w1b = (__bf16*)(ws + off); off += (size_t)(DI * DH) / 2; align8();
  __bf16* w2b = (__bf16*)(ws + off); off += (size_t)(DH * DO) / 2; align8();

  const int T = 256;
  auto nb = [](long long n) { return (unsigned)((n + 255) / 256); };

  // ---- GraphNorm 1 (on x, D=64) ----
  k_zero<<<nb(NGRAPH), T, 0, stream>>>(cnt, NGRAPH);
  k_zero<<<nb((long long)NGRAPH * DI), T, 0, stream>>>(sum, (long long)NGRAPH * DI);
  k_count<<<nb(N), T, 0, stream>>>(batch, cnt, N);
  k_segsum4<<<nb((long long)N * DI / 4), T, 0, stream>>>(x, batch, sum, N, DI);
  k_zero<<<nb((long long)NGRAPH * DI), T, 0, stream>>>(var, (long long)NGRAPH * DI);
  k_center_var4<<<nb((long long)N * DI / 4), T, 0, stream>>>(x, batch, sum, cnt, gn1ms, gn1, var, N, DI);
  k_gn_final4<<<nb((long long)N * DI / 4), T, 0, stream>>>(gn1, batch, var, cnt, gn1w, gn1b, N, DI);

  // ---- GEMM 1: h1 = gn1 @ W1  (WMMA bf16 -> f32), 4 N-tiles per wave ----
  k_pack_b<<<nb(DI * DH), T, 0, stream>>>(W1, w1b, DI, DH);
  {
    long long waves = (long long)((N + 15) / 16) * ((DH / 16) / 4);
    k_gemm_wmma<64, 128, 4><<<nb(waves * 32), T, 0, stream>>>(gn1, w1b, h1, N);
  }

  // ---- degree / symmetric normalization (shared by both convs) ----
  k_deg_init<<<nb(N), T, 0, stream>>>(deg, N);
  k_deg_count<<<nb(E), T, 0, stream>>>(ei, deg, E);
  k_dis<<<nb(N), T, 0, stream>>>(deg, dis, N);

  // ---- GCNConv 1 aggregate + ReLU ----
  k_agg_init4<<<nb((long long)N * DH / 4), T, 0, stream>>>(h1, dis, b1, ag1, N, DH);
  k_agg_edges4<<<nb((long long)E * DH / 4), T, 0, stream>>>(h1, ei, dis, ag1, E, DH);
  k_relu4<<<nb((long long)N * DH / 4), T, 0, stream>>>(ag1, (long long)N * DH / 4);

  // ---- GraphNorm 2 (on ag1, D=128) ----
  k_zero<<<nb((long long)NGRAPH * DH), T, 0, stream>>>(sum, (long long)NGRAPH * DH);
  k_segsum4<<<nb((long long)N * DH / 4), T, 0, stream>>>(ag1, batch, sum, N, DH);
  k_zero<<<nb((long long)NGRAPH * DH), T, 0, stream>>>(var, (long long)NGRAPH * DH);
  k_center_var4<<<nb((long long)N * DH / 4), T, 0, stream>>>(ag1, batch, sum, cnt, gn2ms, gn2, var, N, DH);
  k_gn_final4<<<nb((long long)N * DH / 4), T, 0, stream>>>(gn2, batch, var, cnt, gn2w, gn2b, N, DH);

  // ---- GEMM 2: h2 = gn2 @ W2, 4 N-tiles per wave ----
  k_pack_b<<<nb(DH * DO), T, 0, stream>>>(W2, w2b, DH, DO);
  {
    long long waves = (long long)((N + 15) / 16) * ((DO / 16) / 4);
    k_gemm_wmma<128, 64, 4><<<nb(waves * 32), T, 0, stream>>>(gn2, w2b, h2, N);
  }

  // ---- GCNConv 2 aggregate ----
  k_agg_init4<<<nb((long long)N * DO / 4), T, 0, stream>>>(h2, dis, b2, ag2, N, DO);
  k_agg_edges4<<<nb((long long)E * DO / 4), T, 0, stream>>>(h2, ei, dis, ag2, E, DO);

  // ---- global mean pool -> d_out [256, 64] ----
  k_zero<<<nb((long long)NGRAPH * DO), T, 0, stream>>>(sum, (long long)NGRAPH * DO);
  k_segsum4<<<nb((long long)N * DO / 4), T, 0, stream>>>(ag2, batch, sum, N, DO);
  k_pool_div<<<nb(NGRAPH * DO), T, 0, stream>>>(sum, cnt, out, DO);
}
